// TemporalHierarchicalGoldPredictor_14851996910057
// MI455X (gfx1250) — matrix operand order repaired and checked by reading, not compiled, not run on hardware
//
#include <hip/hip_runtime.h>
#include <stdint.h>

// ---------------- types ----------------
typedef __attribute__((ext_vector_type(16))) _Float16 v16h;
typedef __attribute__((ext_vector_type(8)))  float    v8f;

union V16 { v16h h; uint32_t u[8]; };

// explicit global (address_space(1)) pointers: force global_load/global_store,
// tracked only by LOADcnt/STOREcnt (flat loads also tie up DScnt).
typedef __attribute__((address_space(1))) const float    * gc_float;
typedef __attribute__((address_space(1))) const int      * gc_int;
typedef __attribute__((address_space(1))) const uint32_t * gc_u32;
typedef __attribute__((address_space(1))) float          * g_float;

__device__ inline gc_float gcf(const float* p)   { return (gc_float)(uintptr_t)p; }
__device__ inline gc_int   gci(const int* p)     { return (gc_int)(uintptr_t)p; }
__device__ inline gc_u32   gcu(const uint32_t* p){ return (gc_u32)(uintptr_t)p; }
__device__ inline g_float  gf (float* p)         { return (g_float)(uintptr_t)p; }

// fast sigmoid / tanh: single v_exp + single v_rcp (no IEEE div expansion)
__device__ inline float sigm(float x) {
  return __builtin_amdgcn_rcpf(1.f + __expf(-x));
}
__device__ inline float tanh_(float x) {
  // tanh(x) = 2*sigm(2x) - 1
  return fmaf(2.f, __builtin_amdgcn_rcpf(1.f + __expf(-2.f * x)), -1.f);
}
__device__ inline float leaky(float x) { return x > 0.f ? x : 0.2f * x; }

// pack two f32 -> one dword of f16x2 with a single v_cvt_pk_rtz_f16_f32
__device__ inline uint32_t pk2(float a, float b) {
  auto r = __builtin_amdgcn_cvt_pkrtz(a, b);  // __fp16 ext_vector(2)
  return __builtin_bit_cast(uint32_t, r);
}

// lane L <-> lane L^16 swap on wave32 via v_permlanex16_b32 (VALU, no DS pipe)
__device__ inline uint32_t xswap(uint32_t v) {
  return (uint32_t)__builtin_amdgcn_permlanex16((int)v, (int)v, 0x76543210, 0xfedcba98,
                                                false, false);
}
__device__ inline float xswapf(float v) {
  return __builtin_bit_cast(float, xswap(__builtin_bit_cast(uint32_t, v)));
}

// ---------------- input index map (JAX pytree: dicts flatten with sorted keys) ----------------
enum {
  IN_X1 = 0, IN_X2, IN_G1, IN_G2, IN_G3, IN_G4, IN_G5, IN_LENS,
  P_AH_L1W = 8, P_AH_L1B, P_AH_L2W, P_AH_L2B,
  P_AL_WHHB, P_AL_WHHF, P_AL_WIHB, P_AL_WIHF, P_AL_BB, P_AL_BF,
  P_BH_L1W, P_BH_L1B, P_BH_L2W, P_BH_L2B,
  P_BL_WHHB, P_BL_WHHF, P_BL_WIHB, P_BL_WIHF, P_BL_BB, P_BL_BF,
  P_C_L1W, P_C_L1B, P_C_L2W, P_C_L2B, P_C_L3W, P_C_L3B,
  P_D_KW, P_D_KB, P_D_O1W, P_D_O1B, P_D_O2W, P_D_O2B, P_D_QW, P_D_QB, P_D_VW, P_D_VB,
  P_E_KW, P_E_KB, P_E_O1W, P_E_O1B, P_E_O2W, P_E_O2B, P_E_QW, P_E_QB, P_E_VW, P_E_VB,
  P_F_L1W, P_F_L1B, P_F_L2W, P_F_L2B,
  P_G1H_W, P_G1H_B, P_G1L_WHHB, P_G1L_WHHF, P_G1L_WIHB, P_G1L_WIHF, P_G1L_BB, P_G1L_BF,
  P_G2H_W, P_G2H_B, P_G2L_WHHB, P_G2L_WHHF, P_G2L_WIHB, P_G2L_WIHF, P_G2L_BB, P_G2L_BF,
  P_G3H_W, P_G3H_B, P_G3L_WHHB, P_G3L_WHHF, P_G3L_WIHB, P_G3L_WIHF, P_G3L_BB, P_G3L_BF,
  P_G4H_W, P_G4H_B, P_G4L_WHHB, P_G4L_WHHF, P_G4L_WIHB, P_G4L_WIHF, P_G4L_BB, P_G4L_BF,
  P_G5H_W, P_G5H_B, P_G5L_WHHB, P_G5L_WHHF, P_G5L_WIHB, P_G5L_WIHF, P_G5L_BB, P_G5L_BF,
  P_FIN_L1W, P_FIN_L1B, P_FIN_L2W, P_FIN_L2B
};

// ---------------- workspace layout ----------------
#define PK_A   0
#define PK_B   4096
#define PK_G1  5120
#define PK_G2  9216
#define PK_G3  10240
#define PK_G4  11264
#define PK_G5  11776
#define PK_TOT 12288
#define HF_A   0
#define HF_B   8192
#define HF_G1  12288
#define HF_G2  94208
#define HF_G3  135168
#define HF_G4  176128
#define HF_G5  196608
#define HF_END 217088
#define HB_BASE 217088
#define GC_OFF  434176
#define PE_OFF  536576
#define T1_OFF  577536
#define T2_OFF  581632
#define DMG_OFF 585728
#define VIS_OFF 585984

// =================================================================
// 1. Weight packing: f16 A-matrix tiles in WMMA register layout.
//    Combined K axis: [Whh (H) | Wih (fin) | bias (1) | pad] -> 32*KB.
// =================================================================
struct PackDesc { const float* Whh; const float* Wih; const float* b; int H; int FIN; int NT; int KB; int off; };
struct PackArgs { PackDesc d[7]; };

__device__ inline float packW(const PackDesc& D, int k, int row) {
  if (k < D.H) return gcf(D.Whh)[row * D.H + k];
  int k2 = k - D.H;
  if (k2 < D.FIN) return gcf(D.Wih)[row * D.FIN + k2];
  if (k2 == D.FIN) return gcf(D.b)[row];
  return 0.f;
}

__global__ void pack_weights_kernel(PackArgs pa, uint32_t* __restrict__ pack) {
  int idx = blockIdx.x * blockDim.x + threadIdx.x;
  if (idx >= PK_TOT) return;
  int l = 0;
  for (; l < 7; ++l) {
    int cnt = pa.d[l].NT * pa.d[l].KB * 256;
    if (idx < cnt) break;
    idx -= cnt;
  }
  const PackDesc D = pa.d[l];
  int v = idx & 7;
  int lane = (idx >> 3) & 31;
  int kbm = idx >> 8;
  int kb = kbm % D.KB;
  int m  = kbm / D.KB;
  int row = 16 * m + (lane & 15);
  bool hiL = lane >= 16;
  // A-matrix 16x32 f16 layout, contiguous k order: k = 16*half + element
  int kk0 = (hiL ? 16 : 0) + 2 * v;
  float a = packW(D, kb * 32 + kk0,     row);
  float b = packW(D, kb * 32 + kk0 + 1, row);
  pack[D.off + ((m * D.KB + kb) * 32 + lane) * 8 + v] = pk2(a, b);
}

// =================================================================
// 2. Forward LSTM scans. One wave32 = 16 sequences. D = W x [h;x;1].
// =================================================================
template <int H, int FIN>
__device__ void lstm_fwd_wave(const float* __restrict__ xin, const int* __restrict__ lensin,
                              int lenDiv, const uint32_t* __restrict__ packin,
                              float* __restrict__ hfOutin, int seqBase) {
  constexpr int NT = (4 * H) / 16;
  constexpr int KB = (H == 32) ? 2 : 1;
  constexpr int NU = (H == 32) ? 16 : 8;
  constexpr int T = 512;

  const int lane = threadIdx.x & 31;
  const bool isLo = lane < 16;
  const int seq = seqBase + (lane & 15);
  const int len = gci(lensin)[seq / lenDiv];
  gc_float xr0 = gcf(xin) + (size_t)seq * T * FIN;
  gc_u32   pack = gcu(packin);

  // weights resident in VGPRs for the whole scan
  V16 wA[NT][KB];
#pragma unroll
  for (int m = 0; m < NT; ++m) {
#pragma unroll
    for (int k = 0; k < KB; ++k) {
#pragma unroll
      for (int v = 0; v < 8; ++v) {
        wA[m][k].u[v] = pack[(((m * KB) + k) * 32 + lane) * 8 + v];
      }
    }
  }

  float hst[NU], cst[NU];
#pragma unroll
  for (int j = 0; j < NU; ++j) { hst[j] = 0.f; cst[j] = 0.f; }

  for (int t = 0; t < T; ++t) {
    // every lane loads its sequence row (both half-waves same row: hits WGP$,
    // avoids per-step EXEC save/restore); lane-role selects below gate the use.
    float feat[16];
#pragma unroll
    for (int f = 0; f < 16; ++f) feat[f] = 0.f;
    {
      gc_float xr = xr0 + t * FIN;
#pragma unroll
      for (int f = 0; f < FIN; ++f) feat[f] = xr[f];
      feat[FIN] = 1.0f;  // bias column
      // branchless speculative prefetch a few steps ahead (clamped in-bounds)
      int tp = t + 4 < T ? t + 4 : T - 1;
      __builtin_prefetch((const float*)(uintptr_t)(xr0 + tp * FIN), 0, 1);
    }

    V16 B0, B1;
    if constexpr (H == 32) {
      uint32_t gA[4], gB[4], sA[4], sB[4];
#pragma unroll
      for (int v = 0; v < 4; ++v) {
        gA[v] = pk2(hst[2 * v], hst[2 * v + 1]);
        gB[v] = pk2(hst[8 + 2 * v], hst[9 + 2 * v]);
      }
#pragma unroll
      for (int v = 0; v < 4; ++v) {
        sA[v] = xswap(gA[v]);
        sB[v] = xswap(gB[v]);
      }
#pragma unroll
      for (int v = 0; v < 4; ++v) {
        B0.u[v]     = isLo ? gA[v] : sB[v];
        B0.u[4 + v] = isLo ? sA[v] : gB[v];
      }
#pragma unroll
      for (int v = 0; v < 8; ++v) B1.u[v] = isLo ? pk2(feat[2 * v], feat[2 * v + 1]) : 0u;
    } else if constexpr (H == 16) {
      uint32_t gh[4], sh[4];
#pragma unroll
      for (int v = 0; v < 4; ++v) gh[v] = pk2(hst[2 * v], hst[2 * v + 1]);
#pragma unroll
      for (int v = 0; v < 4; ++v) sh[v] = xswap(gh[v]);
#pragma unroll
      for (int v = 0; v < 4; ++v) {
        B0.u[v]     = isLo ? gh[v] : pk2(feat[2 * v], feat[2 * v + 1]);
        B0.u[4 + v] = isLo ? sh[v] : pk2(feat[8 + 2 * v], feat[9 + 2 * v]);
      }
    } else {  // H == 8
#pragma unroll
      for (int v = 0; v < 4; ++v) {
        B0.u[v]     = isLo ? pk2(hst[2 * v], hst[2 * v + 1]) : 0u;
        B0.u[4 + v] = isLo ? pk2(feat[2 * v], feat[2 * v + 1]) : 0u;
      }
    }

    v8f acc[NT];
    const v8f zc = {0.f, 0.f, 0.f, 0.f, 0.f, 0.f, 0.f, 0.f};
#pragma unroll
    for (int m = 0; m < NT; ++m) {
      v8f d = __builtin_amdgcn_wmma_f32_16x16x32_f16(false, wA[m][0].h, false, B0.h,
                                                     (short)0, zc, false, false);
      if constexpr (KB == 2) {
        d = __builtin_amdgcn_wmma_f32_16x16x32_f16(false, wA[m][1].h, false, B1.h,
                                                   (short)0, d, false, false);
      }
      acc[m] = d;
    }

    const bool act = (t < len);
    if constexpr (H == 32) {
#pragma unroll
      for (int j = 0; j < 16; ++j) {
        int ti = j >> 3, r = j & 7;
        float zi = acc[0 + ti][r], zf = acc[2 + ti][r], zg = acc[4 + ti][r], zo = acc[6 + ti][r];
        float c2 = sigm(zf) * cst[j] + sigm(zi) * tanh_(zg);
        float h2 = sigm(zo) * tanh_(c2);
        cst[j] = act ? c2 : cst[j];
        hst[j] = act ? h2 : hst[j];
      }
    } else if constexpr (H == 16) {
#pragma unroll
      for (int j = 0; j < 8; ++j) {
        float zi = acc[0][j], zf = acc[1][j], zg = acc[2][j], zo = acc[3][j];
        float c2 = sigm(zf) * cst[j] + sigm(zi) * tanh_(zg);
        float h2 = sigm(zo) * tanh_(c2);
        cst[j] = act ? c2 : cst[j];
        hst[j] = act ? h2 : hst[j];
      }
    } else {
      // H == 8: lanes 0-15 hold (i,g), 16-31 hold (f,o).
      // Branchless: u1 = lo ? tanh(a1) : sigm(a1) via tanh(x)=2*sigm(2x)-1.
      const float sc  = isLo ? 2.f : 1.f;
      const float off = isLo ? -1.f : 0.f;
#pragma unroll
      for (int j = 0; j < 8; ++j) {
        float u0 = sigm(acc[0][j]);                 // lo: sig(i), hi: sig(f)
        float s1 = sigm(acc[1][j] * sc);
        float u1 = fmaf(s1, sc, off);               // lo: tanh(g), hi: sig(o)
        float p0 = xswapf(u0);                      // lo receives sig(f)
        float p1 = xswapf(u1);                      // lo receives sig(o)
        float c2 = p0 * cst[j] + u0 * u1;           // hi-lane result is dead
        float h2 = p1 * tanh_(c2);
        cst[j] = act ? c2 : cst[j];
        hst[j] = act ? h2 : hst[j];
      }
    }
  }

  g_float o = gf(hfOutin) + (size_t)seq * H;
  if constexpr (H == 32) {
#pragma unroll
    for (int j = 0; j < 16; ++j) o[(j < 8 ? j : j + 8) + (isLo ? 0 : 8)] = hst[j];
  } else if constexpr (H == 16) {
#pragma unroll
    for (int j = 0; j < 8; ++j) o[j + (isLo ? 0 : 8)] = hst[j];
  } else {
    if (isLo) {
#pragma unroll
      for (int j = 0; j < 8; ++j) o[j] = hst[j];
    }
  }
}

struct FwdArgs {
  const float* x[7];
  const int* lens;
  const uint32_t* pack;
  float* hf;
};

__global__ __launch_bounds__(32) void lstm_fwd_kernel(FwdArgs a) {
  int g = blockIdx.x;
  if (g < 16)        lstm_fwd_wave<32, 8>(a.x[0], a.lens, 1,  a.pack + PK_A,  a.hf + HF_A,  g * 16);
  else if (g < 32)   lstm_fwd_wave<16, 6>(a.x[1], a.lens, 1,  a.pack + PK_B,  a.hf + HF_B,  (g - 16) * 16);
  else if (g < 192)  lstm_fwd_wave<32, 10>(a.x[2], a.lens, 10, a.pack + PK_G1, a.hf + HF_G1, (g - 32) * 16);
  else if (g < 352)  lstm_fwd_wave<16, 8>(a.x[3], a.lens, 10, a.pack + PK_G2, a.hf + HF_G2, (g - 192) * 16);
  else if (g < 512)  lstm_fwd_wave<16, 4>(a.x[4], a.lens, 10, a.pack + PK_G3, a.hf + HF_G3, (g - 352) * 16);
  else if (g < 672)  lstm_fwd_wave<8, 5>(a.x[5], a.lens, 10, a.pack + PK_G4, a.hf + HF_G4, (g - 512) * 16);
  else               lstm_fwd_wave<8, 4>(a.x[6], a.lens, 10, a.pack + PK_G5, a.hf + HF_G5, (g - 672) * 16);
}

// =================================================================
// 3. Backward direction: single cell step from zero state on x[len-1].
// =================================================================
struct BwdDesc { const float* x; const float* Wih; const float* b; float* out; int H; int FIN; int nUnits; int div; };
struct BwdArgs { BwdDesc d[7]; const int* lens; };

__global__ void lstm_bwd_kernel(BwdArgs a) {
  int idx = blockIdx.x * 256 + threadIdx.x;
  int l = 0;
  for (; l < 7; ++l) {
    if (idx < a.d[l].nUnits) break;
    idx -= a.d[l].nUnits;
  }
  if (l >= 7) return;
  const BwdDesc D = a.d[l];
  gc_float Wih = gcf(D.Wih);
  gc_float bb  = gcf(D.b);
  int seq = idx / D.H, unit = idx % D.H;
  int len = gci(a.lens)[seq / D.div];
  gc_float xr = gcf(D.x) + ((size_t)seq * 512 + (len - 1)) * D.FIN;
  float zi = bb[unit], zg = bb[2 * D.H + unit], zo = bb[3 * D.H + unit];
  for (int f = 0; f < D.FIN; ++f) {
    float xv = xr[f];
    zi += Wih[unit * D.FIN + f] * xv;
    zg += Wih[(2 * D.H + unit) * D.FIN + f] * xv;
    zo += Wih[(3 * D.H + unit) * D.FIN + f] * xv;
  }
  float c2 = sigm(zi) * tanh_(zg);
  gf(D.out)[seq * D.H + unit] = sigm(zo) * tanh_(c2);
}

// =================================================================
// 4. A/B heads (mlp2) -> dmg, vis
// =================================================================
__global__ void head_ab_kernel(const float* hfA, const float* hbA, const float* hfB, const float* hbB,
                               const float* A1W, const float* A1b, const float* A2W, const float* A2b,
                               const float* B1W, const float* B1b, const float* B2W, const float* B2b,
                               float* dmg, float* vis) {
  int n = blockIdx.x, t = threadIdx.x;
  __shared__ float inA[64], hidA[32], inB[32], hidB[16];
  if (t < 32) inA[t] = hfA[n * 32 + t];
  else        inA[t] = hbA[n * 32 + (t - 32)];
  if (t < 16) inB[t] = hfB[n * 16 + t];
  else if (t < 32) inB[t] = hbB[n * 16 + (t - 16)];
  __syncthreads();
  if (t < 32) {
    float s = A1b[t];
    for (int k = 0; k < 64; ++k) s += A1W[t * 64 + k] * inA[k];
    hidA[t] = leaky(s);
  } else if (t < 48) {
    int u = t - 32;
    float s = B1b[u];
    for (int k = 0; k < 32; ++k) s += B1W[u * 32 + k] * inB[k];
    hidB[u] = leaky(s);
  }
  __syncthreads();
  if (t == 0) {
    float s = A2b[0];
    for (int k = 0; k < 32; ++k) s += A2W[k] * hidA[k];
    dmg[n] = s;
  }
  if (t == 1) {
    float s = B2b[0];
    for (int k = 0; k < 16; ++k) s += B2W[k] * hidB[k];
    vis[n] = s;
  }
}

// =================================================================
// 5. G heads (single linear) -> gc [2560, 40]
// =================================================================
struct GHDesc { const float* hf; const float* hb; const float* W; const float* b; int H; };
struct GHArgs { GHDesc d[5]; };

__global__ void g_heads_kernel(GHArgs a, float* __restrict__ gc) {
  int idx = blockIdx.x * 256 + threadIdx.x;
  if (idx >= 2560 * 40) return;
  int o = idx & 7, gi = (idx >> 3) % 5, s = idx / 40;
  const GHDesc D = a.d[gi];
  int H2 = 2 * D.H;
  float acc = gcf(D.b)[o];
  gc_float W  = gcf(D.W);
  gc_float hf = gcf(D.hf);
  gc_float hb = gcf(D.hb);
  for (int k = 0; k < D.H; ++k) acc += W[o * H2 + k] * hf[s * D.H + k];
  for (int k = 0; k < D.H; ++k) acc += W[o * H2 + D.H + k] * hb[s * D.H + k];
  gc[s * 40 + gi * 8 + o] = acc;
}

// =================================================================
// 6. F mlp2: gc[row,40] -> pe[row,16]
// =================================================================
__global__ void f_mlp_kernel(const float* gc, const float* W1, const float* b1,
                             const float* W2, const float* b2, float* pe) {
  int row = blockIdx.x, t = threadIdx.x;
  __shared__ float xin[40], hid[32];
  if (t < 40) xin[t] = gc[row * 40 + t];
  __syncthreads();
  if (t < 32) {
    float s = b1[t];
    for (int k = 0; k < 40; ++k) s += W1[t * 40 + k] * xin[k];
    hid[t] = leaky(s);
  }
  __syncthreads();
  if (t < 16) {
    float s = b2[t];
    for (int k = 0; k < 32; ++k) s += W2[t * 32 + k] * hid[k];
    pe[row * 16 + t] = s;
  }
}

// =================================================================
// 7. Attention pooling (D on pe[:, :5], E on pe[:, 5:])
// =================================================================
struct AttnP { const float *qW, *qb, *kW, *kb, *vW, *vb, *o1W, *o1b, *o2W, *o2b; };

__global__ void attn_kernel(const float* pe, AttnP pD, AttnP pE, float* t1, float* t2) {
  int n = blockIdx.x;
  int which = blockIdx.y;
  AttnP P = which ? pE : pD;
  float* tout = which ? t2 : t1;
  int tid = threadIdx.x;  // blockDim = 80
  int p = tid >> 4, d = tid & 15;
  __shared__ float X[5][16], Q[5][16], Kk[5][16], V[5][16], S[5][5], Wm[5][5], pooled[16], hid[32];
  if (tid < 80) X[p][d] = pe[(n * 10 + which * 5 + p) * 16 + d];
  __syncthreads();
  if (tid < 80) {
    float q = P.qb[d], k = P.kb[d], v = P.vb[d];
    for (int c = 0; c < 16; ++c) {
      float xv = X[p][c];
      q += P.qW[d * 16 + c] * xv;
      k += P.kW[d * 16 + c] * xv;
      v += P.vW[d * 16 + c] * xv;
    }
    Q[p][d] = q; Kk[p][d] = k; V[p][d] = v;
  }
  __syncthreads();
  if (tid < 25) {
    int pp = tid / 5, qq = tid % 5;
    float s = 0.f;
    for (int c = 0; c < 16; ++c) s += Q[pp][c] * Kk[qq][c];
    S[pp][qq] = s * 0.25f;  // / sqrt(16)
  }
  __syncthreads();
  if (tid < 5) {
    float mx = S[tid][0];
    for (int q2 = 1; q2 < 5; ++q2) mx = fmaxf(mx, S[tid][q2]);
    float e[5], sm = 0.f;
    for (int q2 = 0; q2 < 5; ++q2) { e[q2] = __expf(S[tid][q2] - mx); sm += e[q2]; }
    float r = __builtin_amdgcn_rcpf(sm);
    for (int q2 = 0; q2 < 5; ++q2) Wm[tid][q2] = e[q2] * r;
  }
  __syncthreads();
  if (tid < 16) {
    float s = 0.f;
    for (int pp = 0; pp < 5; ++pp)
      for (int qq = 0; qq < 5; ++qq) s += Wm[pp][qq] * V[qq][tid];
    pooled[tid] = s * 0.2f;  // mean over p
  }
  __syncthreads();
  if (tid < 32) {
    float s = P.o1b[tid];
    for (int c = 0; c < 16; ++c) s += P.o1W[tid * 16 + c] * pooled[c];
    hid[tid] = leaky(s);
  }
  __syncthreads();
  if (tid < 16) {
    float s = P.o2b[tid];
    for (int c = 0; c < 32; ++c) s += P.o2W[tid * 32 + c] * hid[c];
    tout[n * 16 + tid] = s;
  }
}

// =================================================================
// 8. C tower + final MLP -> out[256]
// =================================================================
__global__ void c_final_kernel(const float* t1, const float* t2, const float* dmg, const float* vis,
                               const float* C1W, const float* C1b, const float* C2W, const float* C2b,
                               const float* C3W, const float* C3b,
                               const float* F1W, const float* F1b, const float* F2W, const float* F2b,
                               float* out) {
  int n = blockIdx.x, t = threadIdx.x;
  __shared__ float tc[32], h1[64], h2[32], comb[3], fh[32];
  if (t < 16) tc[t] = t1[n * 16 + t];
  else if (t < 32) tc[t] = t2[n * 16 + (t - 16)];
  __syncthreads();
  if (t < 64) {
    float s = C1b[t];
    for (int k = 0; k < 32; ++k) s += C1W[t * 32 + k] * tc[k];
    h1[t] = leaky(s);
  }
  __syncthreads();
  if (t < 32) {
    float s = C2b[t];
    for (int k = 0; k < 64; ++k) s += C2W[t * 64 + k] * h1[k];
    h2[t] = leaky(s);
  }
  __syncthreads();
  if (t == 0) {
    float s = C3b[0];
    for (int k = 0; k < 32; ++k) s += C3W[k] * h2[k];
    comb[0] = dmg[n]; comb[1] = vis[n]; comb[2] = s;
  }
  __syncthreads();
  if (t < 32) {
    float s = F1b[t];
    for (int k = 0; k < 3; ++k) s += F1W[t * 3 + k] * comb[k];
    fh[t] = leaky(s);
  }
  __syncthreads();
  if (t == 0) {
    float s = F2b[0];
    for (int k = 0; k < 32; ++k) s += F2W[k] * fh[k];
    out[n] = s;
  }
}

// =================================================================
// launch
// =================================================================
extern "C" void kernel_launch(void* const* d_in, const int* in_sizes, int n_in,
                              void* d_out, int out_size, void* d_ws, size_t ws_size,
                              hipStream_t stream) {
  (void)in_sizes; (void)n_in; (void)out_size; (void)ws_size;
  auto PF = [&](int i) { return (const float*)d_in[i]; };
  const int* lens = (const int*)d_in[IN_LENS];

  uint32_t* pk = (uint32_t*)d_ws;
  float* wf = (float*)(pk + PK_TOT);

  // 1) pack LSTM forward weights into WMMA A-operand layout (bias folded in)
  PackArgs pa;
  auto setP = [&](int i, int whh, int wih, int bb, int H, int FIN, int off) {
    pa.d[i] = {PF(whh), PF(wih), PF(bb), H, FIN, (4 * H) / 16, (H == 32) ? 2 : 1, off};
  };
  setP(0, P_AL_WHHF, P_AL_WIHF, P_AL_BF, 32, 8,  PK_A);
  setP(1, P_BL_WHHF, P_BL_WIHF, P_BL_BF, 16, 6,  PK_B);
  setP(2, P_G1L_WHHF, P_G1L_WIHF, P_G1L_BF, 32, 10, PK_G1);
  setP(3, P_G2L_WHHF, P_G2L_WIHF, P_G2L_BF, 16, 8,  PK_G2);
  setP(4, P_G3L_WHHF, P_G3L_WIHF, P_G3L_BF, 16, 4,  PK_G3);
  setP(5, P_G4L_WHHF, P_G4L_WIHF, P_G4L_BF, 8,  5,  PK_G4);
  setP(6, P_G5L_WHHF, P_G5L_WIHF, P_G5L_BF, 8,  4,  PK_G5);
  pack_weights_kernel<<<48, 256, 0, stream>>>(pa, pk);

  // 2) forward scans (WMMA), 832 independent waves of 16 sequences
  FwdArgs fa;
  fa.x[0] = PF(IN_X1); fa.x[1] = PF(IN_X2);
  fa.x[2] = PF(IN_G1); fa.x[3] = PF(IN_G2); fa.x[4] = PF(IN_G3);
  fa.x[5] = PF(IN_G4); fa.x[6] = PF(IN_G5);
  fa.lens = lens; fa.pack = pk; fa.hf = wf;
  lstm_fwd_kernel<<<832, 32, 0, stream>>>(fa);

  // 3) backward single-step cells
  BwdArgs ba;
  float* hb = wf + HB_BASE;
  auto setB = [&](int i, int x, int wih, int bb, int H, int FIN, int nseq, int div, int off) {
    ba.d[i] = {PF(x), PF(wih), PF(bb), hb + off, H, FIN, nseq * H, div};
  };
  setB(0, IN_X1, P_AL_WIHB, P_AL_BB, 32, 8, 256, 1, HF_A);
  setB(1, IN_X2, P_BL_WIHB, P_BL_BB, 16, 6, 256, 1, HF_B);
  setB(2, IN_G1, P_G1L_WIHB, P_G1L_BB, 32, 10, 2560, 10, HF_G1);
  setB(3, IN_G2, P_G2L_WIHB, P_G2L_BB, 16, 8, 2560, 10, HF_G2);
  setB(4, IN_G3, P_G3L_WIHB, P_G3L_BB, 16, 4, 2560, 10, HF_G3);
  setB(5, IN_G4, P_G4L_WIHB, P_G4L_BB, 8, 5, 2560, 10, HF_G4);
  setB(6, IN_G5, P_G5L_WIHB, P_G5L_BB, 8, 4, 2560, 10, HF_G5);
  ba.lens = lens;
  lstm_bwd_kernel<<<(HF_END + 255) / 256, 256, 0, stream>>>(ba);

  // 4) A/B heads
  head_ab_kernel<<<256, 64, 0, stream>>>(
      wf + HF_A, hb + HF_A, wf + HF_B, hb + HF_B,
      PF(P_AH_L1W), PF(P_AH_L1B), PF(P_AH_L2W), PF(P_AH_L2B),
      PF(P_BH_L1W), PF(P_BH_L1B), PF(P_BH_L2W), PF(P_BH_L2B),
      wf + DMG_OFF, wf + VIS_OFF);

  // 5) G heads -> gc
  GHArgs ga;
  ga.d[0] = {wf + HF_G1, hb + HF_G1, PF(P_G1H_W), PF(P_G1H_B), 32};
  ga.d[1] = {wf + HF_G2, hb + HF_G2, PF(P_G2H_W), PF(P_G2H_B), 16};
  ga.d[2] = {wf + HF_G3, hb + HF_G3, PF(P_G3H_W), PF(P_G3H_B), 16};
  ga.d[3] = {wf + HF_G4, hb + HF_G4, PF(P_G4H_W), PF(P_G4H_B), 8};
  ga.d[4] = {wf + HF_G5, hb + HF_G5, PF(P_G5H_W), PF(P_G5H_B), 8};
  g_heads_kernel<<<(2560 * 40 + 255) / 256, 256, 0, stream>>>(ga, wf + GC_OFF);

  // 6) F mlp2 -> pe
  f_mlp_kernel<<<2560, 64, 0, stream>>>(wf + GC_OFF, PF(P_F_L1W), PF(P_F_L1B),
                                        PF(P_F_L2W), PF(P_F_L2B), wf + PE_OFF);

  // 7) attention pools D and E -> t1, t2
  AttnP pD = {PF(P_D_QW), PF(P_D_QB), PF(P_D_KW), PF(P_D_KB), PF(P_D_VW), PF(P_D_VB),
              PF(P_D_O1W), PF(P_D_O1B), PF(P_D_O2W), PF(P_D_O2B)};
  AttnP pE = {PF(P_E_QW), PF(P_E_QB), PF(P_E_KW), PF(P_E_KB), PF(P_E_VW), PF(P_E_VB),
              PF(P_E_O1W), PF(P_E_O1B), PF(P_E_O2W), PF(P_E_O2B)};
  attn_kernel<<<dim3(256, 2), 80, 0, stream>>>(wf + PE_OFF, pD, pE, wf + T1_OFF, wf + T2_OFF);

  // 8) C tower + final -> out
  c_final_kernel<<<256, 64, 0, stream>>>(
      wf + T1_OFF, wf + T2_OFF, wf + DMG_OFF, wf + VIS_OFF,
      PF(P_C_L1W), PF(P_C_L1B), PF(P_C_L2W), PF(P_C_L2B), PF(P_C_L3W), PF(P_C_L3B),
      PF(P_FIN_L1W), PF(P_FIN_L1B), PF(P_FIN_L2W), PF(P_FIN_L2B),
      (float*)d_out);
}
